// Grouped_GEMM_MoE_43765716747083
// MI455X (gfx1250) — matrix operand order repaired and checked by reading, not compiled, not run on hardware
//
#include <hip/hip_runtime.h>
#include <hip/hip_fp16.h>

#define HIDDEN 1024
#define FFN    2816
#define NE     8
#define TOKENS 4096
#define TOPK   2
#define CAP    (TOKENS * TOPK)   // worst case: one expert takes every assignment
#define TM     32                // tokens per workgroup tile
#define FB     256               // FFN chunk width per phase (16 waves x 16 cols)
#define NTH    512               // 16 waves (wave32)
#define NTILE  4                 // 16x16 output column tiles per wave (4*16*16waves=1024)

typedef _Float16 v16h __attribute__((ext_vector_type(16)));
typedef _Float16 v8h  __attribute__((ext_vector_type(8)));
typedef float    v8f  __attribute__((ext_vector_type(8)));

static __device__ __forceinline__ v8f vzero8() {
  v8f z = {0.f, 0.f, 0.f, 0.f, 0.f, 0.f, 0.f, 0.f};
  return z;
}

static __device__ __forceinline__ v8f wmma_f16(v16h a, v16h b, v8f c) {
  // D = A(16x32 f16) * B(32x16 f16) + C(16x16 f32)
  return __builtin_amdgcn_wmma_f32_16x16x32_f16(false, a, false, b, (short)0, c,
                                                false, false);
}

// A-fragment (16x32, f16): lane half selects K base {0,16}; VGPRs 0-3 hold
// K=base..base+7, VGPRs 4-7 hold K=base+8..base+15 (two 16-byte runs).
static __device__ __forceinline__ v16h load_a_frag(const _Float16* rowbase,
                                                   int k, int kA) {
  v16h r;
  *((v8h*)&r)     = *(const v8h*)(rowbase + k + kA);
  *((v8h*)&r + 1) = *(const v8h*)(rowbase + k + kA + 8);
  return r;
}

// ---------------------------------------------------------------- utilities
__global__ void k_zero_f32(float* __restrict__ p, int n) {
  int i = blockIdx.x * blockDim.x + threadIdx.x;
  int s = gridDim.x * blockDim.x;
  for (; i < n; i += s) p[i] = 0.f;
}

__global__ void k_zero_i32(int* __restrict__ p, int n) {
  int i = blockIdx.x * blockDim.x + threadIdx.x;
  if (i < n) p[i] = 0;
}

__global__ void k_cvt_f16(_Float16* __restrict__ dst, const float* __restrict__ src,
                          int n) {
  int i = blockIdx.x * blockDim.x + threadIdx.x;
  int s = gridDim.x * blockDim.x;
  for (; i < n; i += s) dst[i] = (_Float16)src[i];
}

// w2[e][f][h] (f32) -> w2t[e][h][f] (f16), LDS-tiled transpose
__global__ void k_w2_transpose(_Float16* __restrict__ dst,
                               const float* __restrict__ src) {
  __shared__ _Float16 tile[32][33];
  const int e  = blockIdx.z;
  const int h0 = blockIdx.x * 32;
  const int f0 = blockIdx.y * 32;
  const float* s = src + (size_t)e * FFN * HIDDEN;
  _Float16*    d = dst + (size_t)e * HIDDEN * FFN;
#pragma unroll
  for (int i = 0; i < 32; i += 8) {
    int f = f0 + threadIdx.y + i;
    tile[threadIdx.y + i][threadIdx.x] =
        (_Float16)s[(size_t)f * HIDDEN + h0 + threadIdx.x];
  }
  __syncthreads();
#pragma unroll
  for (int i = 0; i < 32; i += 8) {
    int h = h0 + threadIdx.y + i;
    d[(size_t)h * FFN + f0 + threadIdx.x] = tile[threadIdx.x][threadIdx.y + i];
  }
}

__global__ void k_route(const int* __restrict__ top, const float* __restrict__ wts,
                        int* __restrict__ cnt, int* __restrict__ idx,
                        float* __restrict__ gts) {
  int i = blockIdx.x * blockDim.x + threadIdx.x;
  if (i < TOKENS * TOPK) {
    int t = i / TOPK;
    int e = top[i];
    int p = atomicAdd(&cnt[e], 1);
    idx[e * CAP + p] = t;
    gts[e * CAP + p] = wts[i];
  }
}

// ------------------------------------------------------------- fused MoE
__global__ void __launch_bounds__(NTH)
k_moe(const _Float16* __restrict__ xh,   // [TOKENS][HIDDEN] f16
      const _Float16* __restrict__ w1h,  // [NE][2*FFN][HIDDEN] f16 (h contiguous)
      const _Float16* __restrict__ w2t,  // [NE][HIDDEN][FFN]  f16 (f contiguous)
      const int* __restrict__ cnt, const int* __restrict__ idx,
      const float* __restrict__ gts, float* __restrict__ out) {
  constexpr int TILES = CAP / TM;  // 256
  const int e    = blockIdx.x / TILES;
  const int tile = blockIdx.x % TILES;
  const int n    = cnt[e];
  const int m0   = tile * TM;
  if (m0 >= n) return;

  // Padded strides: row starts advance 4 dwords/bank per row -> b128 fragment
  // reads across the 16 rows touch all 64 banks conflict-free.
  __shared__ __align__(32) _Float16 xs[TM][HIDDEN + 8];  // 66,048 B
  __shared__ __align__(32) _Float16 acts[TM][FB + 8];    // 16,896 B
  __shared__ int   toks[TM];
  __shared__ float sg[TM];

  if (threadIdx.x < TM) {
    int  r = threadIdx.x;
    bool v = (m0 + r) < n;
    toks[r] = v ? idx[e * CAP + m0 + r] : 0;
    sg[r]   = v ? gts[e * CAP + m0 + r] : 0.f;
  }
  __syncthreads();

  // Cooperatively stage the 32 x 1024 f16 x-tile (row-gathered, b128 chunks).
  for (int c = threadIdx.x; c < TM * (HIDDEN / 8); c += NTH) {
    int r   = c >> 7;            // HIDDEN/8 = 128 chunks per row
    int off = (c & 127) * 8;
    *(v8h*)&xs[r][off] = *(const v8h*)(xh + (size_t)toks[r] * HIDDEN + off);
  }
  __syncthreads();

  const int wv  = threadIdx.x >> 5;        // 0..15
  const int ln  = threadIdx.x & 31;
  const int nco = ln & 15;                 // column within 16 (B/C layout)
  const int kB  = (ln < 16) ? 0 : 16;      // B fragment K half
  const int kA  = (ln < 16) ? 0 : 16;      // A fragment K base

  const _Float16* w1e = w1h + (size_t)e * (2 * FFN) * HIDDEN;
  const _Float16* w2e = w2t + (size_t)e * HIDDEN * FFN;

  v8f acc[NTILE][2];  // 4 N-tiles (this wave's 64 output cols) x 2 M-subtiles
#pragma unroll
  for (int t = 0; t < NTILE; t++) { acc[t][0] = vzero8(); acc[t][1] = vzero8(); }

  for (int fc = 0; fc < FFN; fc += FB) {   // 11 chunks
    // ---- phase 1: this wave computes 16 f-columns of a,b over K=HIDDEN ----
    const int fw = fc + wv * 16;
    v8f ca0 = vzero8(), ca1 = vzero8(), cb0 = vzero8(), cb1 = vzero8();
    const _Float16* bArow = w1e + (size_t)(fw + nco) * HIDDEN + kB;
    const _Float16* bBrow = w1e + (size_t)(FFN + fw + nco) * HIDDEN + kB;
    for (int k = 0; k < HIDDEN; k += 32) {
      v16h A0 = load_a_frag(&xs[nco][0], k, kA);
      v16h A1 = load_a_frag(&xs[nco + 16][0], k, kA);
      v16h Ba = *(const v16h*)(bArow + k);
      v16h Bb = *(const v16h*)(bBrow + k);
      ca0 = wmma_f16(A0, Ba, ca0);
      ca1 = wmma_f16(A1, Ba, ca1);
      cb0 = wmma_f16(A0, Bb, cb0);
      cb1 = wmma_f16(A1, Bb, cb1);
    }
    __syncthreads();  // previous chunk's phase-2 readers are done with acts
    // SwiGLU, store f16 activations. C layout: lane holds col nco, VGPR v -> row.
    const int colL = wv * 16 + nco;
#pragma unroll
    for (int v = 0; v < 8; v++) {
      const int r = v + ((ln < 16) ? 0 : 8);
      float a0 = ca0[v], b0 = cb0[v];
      acts[r][colL] = (_Float16)((a0 / (1.f + __expf(-a0))) * b0);
      float a1 = ca1[v], b1 = cb1[v];
      acts[r + 16][colL] = (_Float16)((a1 / (1.f + __expf(-a1))) * b1);
    }
    __syncthreads();
    // ---- phase 2: y += act(32 x FB) @ w2t, this wave's 64 output cols ----
    const int h0 = wv * (16 * NTILE);
    for (int kk = 0; kk < FB; kk += 32) {
      v16h A0 = load_a_frag(&acts[nco][0], kk, kA);
      v16h A1 = load_a_frag(&acts[nco + 16][0], kk, kA);
#pragma unroll
      for (int nt = 0; nt < NTILE; nt++) {
        const _Float16* brow =
            w2e + (size_t)(h0 + nt * 16 + nco) * FFN + (fc + kk + kB);
        v16h B = *(const v16h*)brow;
        acc[nt][0] = wmma_f16(A0, B, acc[nt][0]);
        acc[nt][1] = wmma_f16(A1, B, acc[nt][1]);
      }
    }
  }

  // ---- epilogue: gate-weighted scatter-add into out[token][h] ----
#pragma unroll
  for (int nt = 0; nt < NTILE; nt++) {
    const int h = wv * (16 * NTILE) + nt * 16 + nco;
#pragma unroll
    for (int v = 0; v < 8; v++) {
      const int r0 = v + ((ln < 16) ? 0 : 8);
      const int r1 = r0 + 16;
      if (m0 + r0 < n)
        unsafeAtomicAdd(&out[(size_t)toks[r0] * HIDDEN + h], sg[r0] * acc[nt][0][v]);
      if (m0 + r1 < n)
        unsafeAtomicAdd(&out[(size_t)toks[r1] * HIDDEN + h], sg[r1] * acc[nt][1][v]);
    }
  }
}

// ---------------------------------------------------------------- launcher
extern "C" void kernel_launch(void* const* d_in, const int* in_sizes, int n_in,
                              void* d_out, int out_size, void* d_ws, size_t ws_size,
                              hipStream_t stream) {
  const float* x  = (const float*)d_in[0];  // [TOKENS][HIDDEN]
  const float* ew = (const float*)d_in[1];  // [TOKENS][TOPK]
  const int*   te = (const int*)d_in[2];    // [TOKENS][TOPK]
  const float* w1 = (const float*)d_in[3];  // [NE][2*FFN][HIDDEN]
  const float* w2 = (const float*)d_in[4];  // [NE][FFN][HIDDEN]
  float* out = (float*)d_out;               // [TOKENS][HIDDEN]

  char* ws = (char*)d_ws;
  constexpr size_t SZ_XH  = (size_t)TOKENS * HIDDEN * 2;
  constexpr size_t SZ_W1  = (size_t)NE * 2 * FFN * HIDDEN * 2;
  constexpr size_t SZ_W2T = (size_t)NE * HIDDEN * FFN * 2;
  _Float16* xh  = (_Float16*)(ws);
  _Float16* w1h = (_Float16*)(ws + SZ_XH);
  _Float16* w2t = (_Float16*)(ws + SZ_XH + SZ_W1);
  size_t o = SZ_XH + SZ_W1 + SZ_W2T;
  int* cnt = (int*)(ws + o);  o += 64;                       // padded
  int* idx = (int*)(ws + o);  o += (size_t)NE * CAP * 4;
  float* gts = (float*)(ws + o);

  k_zero_f32<<<2048, 256, 0, stream>>>(out, TOKENS * HIDDEN);
  k_zero_i32<<<1, 64, 0, stream>>>(cnt, NE);
  k_cvt_f16<<<4096, 256, 0, stream>>>(xh, x, TOKENS * HIDDEN);
  k_cvt_f16<<<8192, 256, 0, stream>>>(w1h, w1, NE * 2 * FFN * HIDDEN);
  dim3 tg(HIDDEN / 32, FFN / 32, NE);
  k_w2_transpose<<<tg, dim3(32, 8), 0, stream>>>(w2t, w2);
  k_route<<<(TOKENS * TOPK + 255) / 256, 256, 0, stream>>>(te, ew, cnt, idx, gts);
  k_moe<<<NE * (CAP / TM), NTH, 0, stream>>>(xh, w1h, w2t, cnt, idx, gts, out);
}
